// GNNSeedAttention_74723841016250
// MI455X (gfx1250) — matrix operand order repaired and checked by reading, not compiled, not run on hardware
//
#include <hip/hip_runtime.h>
#include <hip/hip_bf16.h>

#define N_NODES_C   131072
#define N_GRAPHS_C  64
#define N_EDGES_C   2097152
#define IN_DIM_C    256
#define HID_C       512
#define LAYERS_C    3
#define BN_EPS_C    1e-5f

typedef __attribute__((ext_vector_type(16))) __bf16 v16bf;
typedef __attribute__((ext_vector_type(8)))  __bf16 v8bf;
typedef __attribute__((ext_vector_type(4)))  __bf16 v4bf;
typedef __attribute__((ext_vector_type(8)))  float  v8f;

// ---------------------------------------------------------------------------
// WMMA fragment loaders (CDNA5 wave32 layouts, CDNA5 ISA §7.12.2)
// A (16x32 bf16, row-major [M,K]): lane%16 = row; half=lane/16.
//   elements 0..7  -> K = k0 + half*8 + 0..7      (contiguous 16B)
//   elements 8..15 -> K = k0 + 16 + half*8 + 0..7 (contiguous 16B)
// B from pre-transposed Bt [N,K] row-major: lane%16 = column;
//   elements 0..7 -> K = k0 + half*8 + 0..7, 8..15 -> K = k0 + 16 + half*8 + 0..7
// ---------------------------------------------------------------------------
__device__ __forceinline__ v16bf frag_a(const __bf16* __restrict__ rowp, int k0, int half) {
  const __bf16* p = rowp + k0 + half * 8;
  v8bf lo = *(const v8bf*)(p);
  v8bf hi = *(const v8bf*)(p + 16);
  return __builtin_shufflevector(lo, hi, 0,1,2,3,4,5,6,7,8,9,10,11,12,13,14,15);
}

__device__ __forceinline__ v16bf frag_b(const __bf16* __restrict__ rowp, int k0, int half) {
  const __bf16* p = rowp + k0 + half * 8;
  v8bf lo = *(const v8bf*)(p);
  v8bf hi = *(const v8bf*)(p + 16);
  return __builtin_shufflevector(lo, hi, 0,1,2,3,4,5,6,7,8,9,10,11,12,13,14,15);
}

// ---------------------------------------------------------------------------
// bf16 WMMA GEMM: C[M,N] = act(A[M,K] @ Bt[N,K]^T + bias)
// EP: 0 = bias+relu -> f32 | 1 = bias+relu -> bf16 | 2 = bias -> f32 | 3 = plain f32
// Wave tile 32(M) x 64(N): 2 A frags x 4 B frags -> 8 WMMAs / 32-K step.
// Block = 256 threads = 8 waves as 2(M) x 4(N) -> 64 x 256 tile.
// grid = (M/64, N/256). K-loop is ping-pong double-buffered (unrolled x2)
// so s_wait_loadcnt covers loads issued a full K-step earlier.
// Requires K % 64 == 0 (K = 256 or 512 here).
// ---------------------------------------------------------------------------
template<int EP>
__global__ void __launch_bounds__(256)
gemm_wmma_bf16(const __bf16* __restrict__ A, const __bf16* __restrict__ Bt,
               const float* __restrict__ bias, float* __restrict__ Cf,
               __bf16* __restrict__ Cb, int M, int N, int K) {
  const int lane = threadIdx.x & 31;
  const int wave = threadIdx.x >> 5;
  const int r    = lane & 15;
  const int half = lane >> 4;
  const int wm   = wave & 1;      // 2 waves along M
  const int wn   = wave >> 1;     // 4 waves along N
  const int m_base = blockIdx.x * 64 + wm * 32;
  const int n_base = blockIdx.y * 256 + wn * 64;

  v8f acc[2][4];
#pragma unroll
  for (int i = 0; i < 2; ++i)
#pragma unroll
    for (int j = 0; j < 4; ++j) acc[i][j] = (v8f)(0.0f);

  const __bf16* arow[2];
  arow[0] = A + (size_t)(m_base + r) * K;
  arow[1] = A + (size_t)(m_base + 16 + r) * K;
  const __bf16* brow[4];
#pragma unroll
  for (int j = 0; j < 4; ++j) brow[j] = Bt + (size_t)(n_base + j * 16 + r) * K;

  // prologue: fill buffer 0 with k=0
  v16bf a0[2], b0[4], a1[2], b1[4];
#pragma unroll
  for (int i = 0; i < 2; ++i) a0[i] = frag_a(arow[i], 0, half);
#pragma unroll
  for (int j = 0; j < 4; ++j) b0[j] = frag_b(brow[j], 0, half);

  for (int k0 = 0; k0 < K; k0 += 64) {
    // near-cache prefetch of the A stream one buffer ahead
    __builtin_prefetch(arow[0] + k0 + 96, 0, 3);
    __builtin_prefetch(arow[1] + k0 + 96, 0, 3);
    // issue loads for k0+32 (always valid: K % 64 == 0)
#pragma unroll
    for (int i = 0; i < 2; ++i) a1[i] = frag_a(arow[i], k0 + 32, half);
#pragma unroll
    for (int j = 0; j < 4; ++j) b1[j] = frag_b(brow[j], k0 + 32, half);
    // compute on buffer 0 (loaded one step earlier)
#pragma unroll
    for (int i = 0; i < 2; ++i)
#pragma unroll
      for (int j = 0; j < 4; ++j)
        acc[i][j] = __builtin_amdgcn_wmma_f32_16x16x32_bf16(
            false, a0[i], false, b0[j], (short)0, acc[i][j], false, false);
    // issue loads for k0+64 into buffer 0
    if (k0 + 64 < K) {
#pragma unroll
      for (int i = 0; i < 2; ++i) a0[i] = frag_a(arow[i], k0 + 64, half);
#pragma unroll
      for (int j = 0; j < 4; ++j) b0[j] = frag_b(brow[j], k0 + 64, half);
    }
    // compute on buffer 1
#pragma unroll
    for (int i = 0; i < 2; ++i)
#pragma unroll
      for (int j = 0; j < 4; ++j)
        acc[i][j] = __builtin_amdgcn_wmma_f32_16x16x32_bf16(
            false, a1[i], false, b1[j], (short)0, acc[i][j], false, false);
  }

#pragma unroll
  for (int i = 0; i < 2; ++i) {
#pragma unroll
    for (int j = 0; j < 4; ++j) {
      const int col = n_base + j * 16 + r;
      const float bv = (EP == 0 || EP == 1 || EP == 2) ? bias[col] : 0.0f;
#pragma unroll
      for (int t = 0; t < 8; ++t) {
        const int row = m_base + i * 16 + half * 8 + t;
        float v = acc[i][j][t] + bv;
        if (EP == 0 || EP == 1) v = fmaxf(v, 0.0f);
        if (EP == 1) Cb[(size_t)row * N + col] = (__bf16)v;
        else         Cf[(size_t)row * N + col] = v;
      }
    }
  }
}

// ---------------------------------------------------------------------------
// helper kernels
// ---------------------------------------------------------------------------
__global__ void zero_f32(float* __restrict__ p, long n) {
  long i = (long)blockIdx.x * blockDim.x + threadIdx.x;
  if (i < n) p[i] = 0.0f;
}
__global__ void zero_i32(int* __restrict__ p, int n) {
  int i = blockIdx.x * blockDim.x + threadIdx.x;
  if (i < n) p[i] = 0;
}

__global__ void f32_to_bf16_k(const float* __restrict__ in, __bf16* __restrict__ out, long n) {
  long i = ((long)blockIdx.x * blockDim.x + threadIdx.x) * 4;
  if (i < n) {
    float4 v = *(const float4*)(in + i);
    v4bf o = {(__bf16)v.x, (__bf16)v.y, (__bf16)v.z, (__bf16)v.w};
    *(v4bf*)(out + i) = o;
  }
}

// W [K,N] row-major f32 -> Wt [N,K] row-major bf16
__global__ void conv_transpose_bf16(const float* __restrict__ W, __bf16* __restrict__ Wt,
                                    int K, int N) {
  int idx = blockIdx.x * blockDim.x + threadIdx.x;
  if (idx < K * N) {
    int k = idx / N, n = idx - k * N;
    Wt[(size_t)n * K + k] = (__bf16)W[idx];
  }
}

// agg[dst] += h[src] over edges; 128 threads (x4 floats) per edge
__global__ void __launch_bounds__(256)
gin_scatter_add(const float* __restrict__ h, const int* __restrict__ src,
                const int* __restrict__ dst, float* __restrict__ agg) {
  long tid = (long)blockIdx.x * 256 + threadIdx.x;
  int e = (int)(tid >> 7);
  int c = (int)(tid & 127) * 4;
  if (e >= N_EDGES_C) return;
  int s = src[e], d = dst[e];
  float4 v = *(const float4*)(h + (size_t)s * HID_C + c);
  float* p = agg + (size_t)d * HID_C + c;
  unsafeAtomicAdd(p + 0, v.x);
  unsafeAtomicAdd(p + 1, v.y);
  unsafeAtomicAdd(p + 2, v.z);
  unsafeAtomicAdd(p + 3, v.w);
}

// z_bf16 = (1+eps[l])*h + agg
__global__ void gin_fuse_eps(const float* __restrict__ h, const float* __restrict__ agg,
                             const float* __restrict__ eps, int l,
                             __bf16* __restrict__ out) {
  long i = ((long)blockIdx.x * blockDim.x + threadIdx.x) * 4;
  float ep = 1.0f + eps[l];
  float4 a = *(const float4*)(h + i);
  float4 b = *(const float4*)(agg + i);
  v4bf o = {(__bf16)(ep * a.x + b.x), (__bf16)(ep * a.y + b.y),
            (__bf16)(ep * a.z + b.z), (__bf16)(ep * a.w + b.w)};
  *(v4bf*)(out + i) = o;
}

// per-channel sum / sumsq partials (block covers 256 nodes, thread = channel)
__global__ void __launch_bounds__(512)
bn_stats(const float* __restrict__ z, float* __restrict__ sums, float* __restrict__ sumsq) {
  int c = threadIdx.x;
  size_t n0 = (size_t)blockIdx.x * 256;
  float s = 0.0f, q = 0.0f;
  for (int i = 0; i < 256; ++i) {
    float v = z[(n0 + i) * HID_C + c];
    s += v; q += v * v;
  }
  unsafeAtomicAdd(&sums[c], s);
  unsafeAtomicAdd(&sumsq[c], q);
}

__global__ void __launch_bounds__(512)
bn_finalize(const float* __restrict__ sums, const float* __restrict__ sumsq,
            const float* __restrict__ g, const float* __restrict__ b, int l,
            float* __restrict__ scale, float* __restrict__ shift) {
  int c = threadIdx.x;
  float inv_n = 1.0f / (float)N_NODES_C;
  float mu  = sums[c] * inv_n;
  float var = sumsq[c] * inv_n - mu * mu;
  float sc  = rsqrtf(var + BN_EPS_C) * g[l * HID_C + c];
  scale[c] = sc;
  shift[c] = b[l * HID_C + c] - mu * sc;
}

__global__ void bn_apply_relu(const float* __restrict__ z, const float* __restrict__ scale,
                              const float* __restrict__ shift, float* __restrict__ hout) {
  long i = ((long)blockIdx.x * blockDim.x + threadIdx.x) * 4;
  int c = (int)(i & (HID_C - 1));
  float4 v = *(const float4*)(z + i);
  float4 o;
  o.x = fmaxf(v.x * scale[c + 0] + shift[c + 0], 0.0f);
  o.y = fmaxf(v.y * scale[c + 1] + shift[c + 1], 0.0f);
  o.z = fmaxf(v.z * scale[c + 2] + shift[c + 2], 0.0f);
  o.w = fmaxf(v.w * scale[c + 3] + shift[c + 3], 0.0f);
  *(float4*)(hout + i) = o;
}

__global__ void count_nodes(const int* __restrict__ gid, int* __restrict__ counts) {
  int n = blockIdx.x * blockDim.x + threadIdx.x;
  if (n < N_NODES_C) atomicAdd(&counts[gid[n]], 1);
}

__global__ void make_offsets(const int* __restrict__ counts, int* __restrict__ offsets) {
  if (threadIdx.x == 0 && blockIdx.x == 0) {
    int acc = 0;
    for (int g = 0; g < N_GRAPHS_C; ++g) { offsets[g] = acc; acc += counts[g]; }
  }
}

__global__ void __launch_bounds__(512)
graph_mean(const float* __restrict__ h, const int* __restrict__ offsets,
           const int* __restrict__ counts, float* __restrict__ mean_g) {
  int g = blockIdx.x, c = threadIdx.x;
  int off = offsets[g], cnt = counts[g];
  float s = 0.0f;
  for (int i = 0; i < cnt; ++i) s += h[(size_t)(off + i) * HID_C + c];
  mean_g[(size_t)g * HID_C + c] = s / fmaxf((float)cnt, 1.0f);
}

__global__ void __launch_bounds__(512)
seed_select(const float* __restrict__ h, const int* __restrict__ seed_idx,
            const int* __restrict__ offsets, const int* __restrict__ counts,
            const float* __restrict__ mean_g, float* __restrict__ hseed) {
  int g = blockIdx.x, c = threadIdx.x;
  int sidx = seed_idx[g], cnt = counts[g], off = offsets[g];
  bool valid = (sidx >= 0) && (sidx < cnt);
  int gi = off + sidx;
  gi = gi < 0 ? 0 : (gi > N_NODES_C - 1 ? N_NODES_C - 1 : gi);
  hseed[(size_t)g * HID_C + c] = valid ? h[(size_t)gi * HID_C + c]
                                       : mean_g[(size_t)g * HID_C + c];
}

// s2[g,n] = sum_k hseed[g,k] * W2[k,n]  (tiny: 64x512x512)
__global__ void __launch_bounds__(512)
seed_w2(const float* __restrict__ hseed, const float* __restrict__ W2,
        float* __restrict__ s2) {
  int g = blockIdx.x, n = threadIdx.x;
  float acc = 0.0f;
  for (int k = 0; k < HID_C; ++k)
    acc += hseed[(size_t)g * HID_C + k] * W2[(size_t)k * HID_C + n];
  s2[(size_t)g * HID_C + n] = acc;
}

// e[n] = leaky(sum_c tanh(zatt[n,c] + s2[gid[n],c]) * w_attn[c]); 1 wave32 per node
__global__ void __launch_bounds__(256)
attn_scores(const float* __restrict__ zatt, const float* __restrict__ s2,
            const float* __restrict__ w_attn, const int* __restrict__ gid,
            float* __restrict__ e) {
  int wid = threadIdx.x >> 5, lane = threadIdx.x & 31;
  int n = blockIdx.x * 8 + wid;
  int g = gid[n];
  float acc = 0.0f;
  for (int c = lane; c < HID_C; c += 32) {
    float t = tanhf(zatt[(size_t)n * HID_C + c] + s2[(size_t)g * HID_C + c]);
    acc += t * w_attn[c];
  }
#pragma unroll
  for (int m = 16; m > 0; m >>= 1) acc += __shfl_xor(acc, m, 32);
  if (lane == 0) e[n] = acc >= 0.0f ? acc : 0.2f * acc;
}

__global__ void __launch_bounds__(256)
seg_softmax_stats(const float* __restrict__ e, const int* __restrict__ offsets,
                  const int* __restrict__ counts, float* __restrict__ emax,
                  float* __restrict__ denom) {
  __shared__ float red[256];
  int g = blockIdx.x, t = threadIdx.x;
  int off = offsets[g], cnt = counts[g];
  float m = -3.4e38f;
  for (int i = t; i < cnt; i += 256) m = fmaxf(m, e[off + i]);
  red[t] = m; __syncthreads();
  for (int s = 128; s > 0; s >>= 1) { if (t < s) red[t] = fmaxf(red[t], red[t + s]); __syncthreads(); }
  m = red[0]; __syncthreads();
  float s = 0.0f;
  for (int i = t; i < cnt; i += 256) s += expf(e[off + i] - m);
  red[t] = s; __syncthreads();
  for (int st = 128; st > 0; st >>= 1) { if (t < st) red[t] += red[t + st]; __syncthreads(); }
  if (t == 0) {
    if (cnt == 0) { emax[g] = 0.0f; denom[g] = 1.0f; }
    else          { emax[g] = m;    denom[g] = red[0]; }
  }
}

__global__ void alpha_kernel(const float* __restrict__ e, const int* __restrict__ gid,
                             const float* __restrict__ emax, const float* __restrict__ denom,
                             float* __restrict__ alpha) {
  int n = blockIdx.x * blockDim.x + threadIdx.x;
  if (n < N_NODES_C) {
    int g = gid[n];
    alpha[n] = expf(e[n] - emax[g]) / denom[g];
  }
}

__global__ void __launch_bounds__(512)
readout_kernel(const float* __restrict__ h, const float* __restrict__ alpha,
               const int* __restrict__ offsets, const int* __restrict__ counts,
               float* __restrict__ ro) {
  int g = blockIdx.x, c = threadIdx.x;
  int off = offsets[g], cnt = counts[g];
  float s = 0.0f;
  for (int i = 0; i < cnt; ++i)
    s += alpha[off + i] * h[(size_t)(off + i) * HID_C + c];
  ro[(size_t)g * HID_C + c] = s;
}

__global__ void __launch_bounds__(512)
logits_kernel(const float* __restrict__ ro, const float* __restrict__ cls_w,
              const float* __restrict__ cls_b, float* __restrict__ out) {
  __shared__ float red[512];
  int g = blockIdx.x, c = threadIdx.x;
  red[c] = ro[(size_t)g * HID_C + c] * cls_w[c];
  __syncthreads();
  for (int s = 256; s > 0; s >>= 1) { if (c < s) red[c] += red[c + s]; __syncthreads(); }
  if (c == 0) out[g] = red[0] + cls_b[0];
}

// ---------------------------------------------------------------------------
extern "C" void kernel_launch(void* const* d_in, const int* in_sizes, int n_in,
                              void* d_out, int out_size, void* d_ws, size_t ws_size,
                              hipStream_t stream) {
  const float* x      = (const float*)d_in[0];
  const float* proj_w = (const float*)d_in[1];
  const float* proj_b = (const float*)d_in[2];
  const float* eps    = (const float*)d_in[3];
  const float* lin1_w = (const float*)d_in[4];
  const float* lin1_b = (const float*)d_in[5];
  const float* lin2_w = (const float*)d_in[6];
  const float* lin2_b = (const float*)d_in[7];
  const float* bn_g   = (const float*)d_in[8];
  const float* bn_b   = (const float*)d_in[9];
  const float* W1     = (const float*)d_in[10];
  const float* W2     = (const float*)d_in[11];
  const float* w_attn = (const float*)d_in[12];
  const float* cls_w  = (const float*)d_in[13];
  const float* cls_b  = (const float*)d_in[14];
  const int*   eidx   = (const int*)d_in[15];
  const int*   gid    = (const int*)d_in[16];
  const int*   seed   = (const int*)d_in[17];
  float* out = (float*)d_out;

  const int* esrc = eidx;
  const int* edst = eidx + N_EDGES_C;

  // --- workspace bump allocator (256B aligned) ---
  char* ws = (char*)d_ws;
  size_t off = 0;
  auto alloc = [&](size_t bytes) -> void* {
    void* p = ws + off;
    off = (off + bytes + 255) & ~(size_t)255;
    return p;
  };
  const size_t NH = (size_t)N_NODES_C * HID_C;
  float*  h      = (float*) alloc(NH * 4);                 // node features f32
  float*  zbuf   = (float*) alloc(NH * 4);                 // agg / z / zatt (reused)
  __bf16* bufA   = (__bf16*)alloc(NH * 2);                 // GEMM A input
  __bf16* bufB   = (__bf16*)alloc(NH * 2);                 // lin1 output
  __bf16* wt_pj  = (__bf16*)alloc((size_t)IN_DIM_C * HID_C * 2);
  __bf16* wt_l1  = (__bf16*)alloc((size_t)LAYERS_C * HID_C * HID_C * 2);
  __bf16* wt_l2  = (__bf16*)alloc((size_t)LAYERS_C * HID_C * HID_C * 2);
  __bf16* wt_W1  = (__bf16*)alloc((size_t)HID_C * HID_C * 2);
  float*  sums   = (float*) alloc(HID_C * 4);
  float*  sumsq  = (float*) alloc(HID_C * 4);
  float*  scale  = (float*) alloc(HID_C * 4);
  float*  shift  = (float*) alloc(HID_C * 4);
  int*    counts = (int*)   alloc(N_GRAPHS_C * 4);
  int*    offs   = (int*)   alloc(N_GRAPHS_C * 4);
  float*  mean_g = (float*) alloc((size_t)N_GRAPHS_C * HID_C * 4);
  float*  hseed  = (float*) alloc((size_t)N_GRAPHS_C * HID_C * 4);
  float*  s2     = (float*) alloc((size_t)N_GRAPHS_C * HID_C * 4);
  float*  evec   = (float*) alloc((size_t)N_NODES_C * 4);
  float*  emax   = (float*) alloc(N_GRAPHS_C * 4);
  float*  denom  = (float*) alloc(N_GRAPHS_C * 4);
  float*  ro     = (float*) alloc((size_t)N_GRAPHS_C * HID_C * 4);
  (void)ws_size; (void)in_sizes; (void)n_in; (void)out_size;

  const dim3 gemm_grid(N_NODES_C / 64, HID_C / 256);   // (2048, 2)
  const int elem_blocks = (int)(NH / 4 / 256);         // 65536 (float4 passes)

  // --- one-time weight conversion to bf16, transposed [N,K] ---
  conv_transpose_bf16<<<(IN_DIM_C * HID_C + 255) / 256, 256, 0, stream>>>(proj_w, wt_pj, IN_DIM_C, HID_C);
  for (int l = 0; l < LAYERS_C; ++l) {
    conv_transpose_bf16<<<(HID_C * HID_C + 255) / 256, 256, 0, stream>>>(
        lin1_w + (size_t)l * HID_C * HID_C, wt_l1 + (size_t)l * HID_C * HID_C, HID_C, HID_C);
    conv_transpose_bf16<<<(HID_C * HID_C + 255) / 256, 256, 0, stream>>>(
        lin2_w + (size_t)l * HID_C * HID_C, wt_l2 + (size_t)l * HID_C * HID_C, HID_C, HID_C);
  }
  conv_transpose_bf16<<<(HID_C * HID_C + 255) / 256, 256, 0, stream>>>(W1, wt_W1, HID_C, HID_C);

  // --- input projection: h = relu(x @ proj_w + b), bf16 WMMA ---
  f32_to_bf16_k<<<(int)(((size_t)N_NODES_C * IN_DIM_C / 4 + 255) / 256), 256, 0, stream>>>(
      x, bufA, (long)N_NODES_C * IN_DIM_C);
  gemm_wmma_bf16<0><<<gemm_grid, 256, 0, stream>>>(bufA, wt_pj, proj_b, h, nullptr,
                                                   N_NODES_C, HID_C, IN_DIM_C);

  // --- GIN layers ---
  for (int l = 0; l < LAYERS_C; ++l) {
    zero_f32<<<(int)((NH + 255) / 256), 256, 0, stream>>>(zbuf, (long)NH);
    gin_scatter_add<<<(int)((size_t)N_EDGES_C * 128 / 256), 256, 0, stream>>>(h, esrc, edst, zbuf);
    gin_fuse_eps<<<elem_blocks, 256, 0, stream>>>(h, zbuf, eps, l, bufA);
    gemm_wmma_bf16<1><<<gemm_grid, 256, 0, stream>>>(
        bufA, wt_l1 + (size_t)l * HID_C * HID_C, lin1_b + (size_t)l * HID_C,
        nullptr, bufB, N_NODES_C, HID_C, HID_C);
    gemm_wmma_bf16<2><<<gemm_grid, 256, 0, stream>>>(
        bufB, wt_l2 + (size_t)l * HID_C * HID_C, lin2_b + (size_t)l * HID_C,
        zbuf, nullptr, N_NODES_C, HID_C, HID_C);
    zero_f32<<<4, 256, 0, stream>>>(sums, HID_C);
    zero_f32<<<4, 256, 0, stream>>>(sumsq, HID_C);
    bn_stats<<<N_NODES_C / 256, 512, 0, stream>>>(zbuf, sums, sumsq);
    bn_finalize<<<1, 512, 0, stream>>>(sums, sumsq, bn_g, bn_b, l, scale, shift);
    bn_apply_relu<<<elem_blocks, 256, 0, stream>>>(zbuf, scale, shift, h);
  }

  // --- per-graph structure ---
  zero_i32<<<1, N_GRAPHS_C, 0, stream>>>(counts, N_GRAPHS_C);
  count_nodes<<<N_NODES_C / 256, 256, 0, stream>>>(gid, counts);
  make_offsets<<<1, 1, 0, stream>>>(counts, offs);
  graph_mean<<<N_GRAPHS_C, 512, 0, stream>>>(h, offs, counts, mean_g);
  seed_select<<<N_GRAPHS_C, 512, 0, stream>>>(h, seed, offs, counts, mean_g, hseed);
  seed_w2<<<N_GRAPHS_C, 512, 0, stream>>>(hseed, W2, s2);

  // --- attention: zatt = h @ W1 (WMMA), then tanh/dot/leaky fused ---
  f32_to_bf16_k<<<elem_blocks, 256, 0, stream>>>(h, bufA, (long)NH);
  gemm_wmma_bf16<3><<<gemm_grid, 256, 0, stream>>>(bufA, wt_W1, nullptr, zbuf, nullptr,
                                                   N_NODES_C, HID_C, HID_C);
  attn_scores<<<N_NODES_C / 8, 256, 0, stream>>>(zbuf, s2, w_attn, gid, evec);

  // --- segment softmax + readout + classifier ---
  seg_softmax_stats<<<N_GRAPHS_C, 256, 0, stream>>>(evec, offs, counts, emax, denom);
  float* alpha_out = out + N_GRAPHS_C;   // output layout: [logits(64) | alpha(131072)]
  alpha_kernel<<<N_NODES_C / 256, 256, 0, stream>>>(evec, gid, emax, denom, alpha_out);
  readout_kernel<<<N_GRAPHS_C, 512, 0, stream>>>(h, alpha_out, offs, counts, ro);
  logits_kernel<<<N_GRAPHS_C, 512, 0, stream>>>(ro, cls_w, cls_b, out);
}
